// MoE_HyperNetLinearLayer_6399501271813
// MI455X (gfx1250) — matrix operand order repaired and checked by reading, not compiled, not run on hardware
//
#include <hip/hip_runtime.h>

#define B_DIM 4096
#define I_DIM 1024
#define O_DIM 1024
#define C_DIM 128
#define E_DIM 8

typedef __attribute__((ext_vector_type(16))) __bf16          v16bf;
typedef __attribute__((ext_vector_type(8)))  float           v8f;
typedef __attribute__((ext_vector_type(4)))  unsigned int    u32x4;
typedef __attribute__((ext_vector_type(4)))  float           f32x4;
typedef __attribute__((ext_vector_type(4)))  unsigned short  u16x4;

union FragAB {
  v16bf v;
  u32x4 u[2];
};

__device__ __forceinline__ unsigned short f2bf(float f) {
  // round-to-nearest-even fp32 -> bf16
  unsigned int u = __float_as_uint(f);
  u += 0x7fffu + ((u >> 16) & 1u);
  return (unsigned short)(u >> 16);
}

// ---------------------------------------------------------------------------
// Kernel 1: gating MLP + softmax.  One thread per batch row.
// ---------------------------------------------------------------------------
__global__ __launch_bounds__(256) void k_gating(
    const float* __restrict__ cond, const float* __restrict__ w1,
    const float* __restrict__ b1, const float* __restrict__ w2,
    const float* __restrict__ b2, float* __restrict__ gating) {
  const int b = blockIdx.x * 256 + threadIdx.x;   // grid sized exactly B/256
  float h[32];
#pragma unroll
  for (int j = 0; j < 32; ++j) h[j] = b1[j];
  const float* cp = cond + (size_t)b * C_DIM;
  for (int c = 0; c < C_DIM; ++c) {
    const float xc = cp[c];
    const float* wr = w1 + (size_t)c * 32;
#pragma unroll
    for (int j = 0; j < 32; ++j) h[j] = __fmaf_rn(xc, wr[j], h[j]);
  }
#pragma unroll
  for (int j = 0; j < 32; ++j) h[j] = fmaxf(h[j], 0.0f);

  float lg[E_DIM];
#pragma unroll
  for (int e = 0; e < E_DIM; ++e) lg[e] = b2[e];
  for (int j = 0; j < 32; ++j) {
    const float hj = h[j];
    const float* wr = w2 + (size_t)j * E_DIM;
#pragma unroll
    for (int e = 0; e < E_DIM; ++e) lg[e] = __fmaf_rn(hj, wr[e], lg[e]);
  }
  float m = lg[0];
#pragma unroll
  for (int e = 1; e < E_DIM; ++e) m = fmaxf(m, lg[e]);
  float s = 0.0f;
#pragma unroll
  for (int e = 0; e < E_DIM; ++e) { lg[e] = __expf(lg[e] - m); s += lg[e]; }
  const float inv = 1.0f / s;
  float* gp = gating + (size_t)b * E_DIM;
#pragma unroll
  for (int e = 0; e < E_DIM; ++e) gp[e] = lg[e] * inv;
}

// ---------------------------------------------------------------------------
// Kernel 2: x fp32 -> bf16, row-major [B, I].  4 elements / thread.
// ---------------------------------------------------------------------------
__global__ __launch_bounds__(256) void k_cvt_x(const float* __restrict__ x,
                                               unsigned short* __restrict__ xb) {
  const int i = blockIdx.x * 256 + threadIdx.x;   // grid sized exactly B*I/4/256
  const f32x4 f = ((const f32x4*)x)[i];
  u16x4 o;
  o.x = f2bf(f.x);
  o.y = f2bf(f.y);
  o.z = f2bf(f.z);
  o.w = f2bf(f.w);
  ((u16x4*)xb)[i] = o;
}

// ---------------------------------------------------------------------------
// Kernel 3: W [E, I, O] fp32 -> Wt [E, O, I] bf16 (LDS-tiled transpose).
// block (32, 8); grid (O/32, I/32, E).
// ---------------------------------------------------------------------------
__global__ __launch_bounds__(256) void k_transpose_w(
    const float* __restrict__ w, unsigned short* __restrict__ wt) {
  __shared__ unsigned short tile[32][33];
  const int e  = blockIdx.z;
  const int o0 = blockIdx.x * 32;
  const int i0 = blockIdx.y * 32;
  const int tx = threadIdx.x;   // 0..31
  const int ty = threadIdx.y;   // 0..7
  const float* wp = w + (size_t)e * I_DIM * O_DIM;
#pragma unroll
  for (int k = 0; k < 32; k += 8)
    tile[ty + k][tx] = f2bf(wp[(size_t)(i0 + ty + k) * O_DIM + o0 + tx]);
  __syncthreads();
  unsigned short* op = wt + (size_t)e * O_DIM * I_DIM;
#pragma unroll
  for (int k = 0; k < 32; k += 8)
    op[(size_t)(o0 + ty + k) * I_DIM + i0 + tx] = tile[tx][ty + k];
}

// ---------------------------------------------------------------------------
// Kernel 4: main WMMA GEMM.
//   out[b,o] = sum_e g[b,e] * (x_bf16[b,:] @ Wt_e^T)[o] + sum_e g[b,e]*bias[e,o]
// 256 threads = 8 waves (wave32), 2x4 wave grid, 64x32 tile per wave,
// 128x128 per workgroup.  #pragma unroll 1 on e/ks loops keeps the loads
// inside the K loop (prevents the unroll+LICM transform that spilled all
// A fragments to scratch).
// ---------------------------------------------------------------------------
__global__ __launch_bounds__(256) void k_moe_gemm(
    const unsigned short* __restrict__ xb,   // [B, I]   bf16, A-matrix
    const unsigned short* __restrict__ wt,   // [E, O, I] bf16, B cols K-contig
    const float* __restrict__ gating,        // [B, E]
    const float* __restrict__ bias,          // [E, O]
    float* __restrict__ out) {               // [B, O]
  const int lane = threadIdx.x & 31;
  const int wave = threadIdx.x >> 5;
  const int wm   = wave >> 2;                 // 0..1  (M direction)
  const int wn   = wave & 3;                  // 0..3  (N direction)
  const int rbase = blockIdx.y * 128 + wm * 64;
  const int cbase = blockIdx.x * 128 + wn * 32;
  const int l16  = lane & 15;
  const int half = lane >> 4;                 // 0 or 1 (K-half select)

  v8f tot[4][2];
#pragma unroll
  for (int mt = 0; mt < 4; ++mt)
#pragma unroll
    for (int nt = 0; nt < 2; ++nt)
#pragma unroll
      for (int v = 0; v < 8; ++v) tot[mt][nt][v] = 0.0f;

  // A fragment base: lane holds row M = l16; two b128 chunks at +0 / +32B,
  // with lane-half K offset of 8 elements.
  const unsigned short* abase =
      xb + (size_t)(rbase + l16) * I_DIM + half * 8;

#pragma unroll 1
  for (int e = 0; e < E_DIM; ++e) {
    // B fragment base: lane holds column N = l16; two b128 chunks at +0 / +32B,
    // lane-half K offset of 8 elements.  Wt row = output column.
    const unsigned short* ap = abase;
    const unsigned short* bp =
        wt + ((size_t)e * O_DIM + (size_t)(cbase + l16)) * I_DIM + half * 8;

    v8f acc[4][2];
#pragma unroll
    for (int mt = 0; mt < 4; ++mt)
#pragma unroll
      for (int nt = 0; nt < 2; ++nt)
#pragma unroll
        for (int v = 0; v < 8; ++v) acc[mt][nt][v] = 0.0f;

#pragma unroll 1
    for (int ks = 0; ks < I_DIM / 32; ++ks) {
      FragAB a[4], bf[2];
#pragma unroll
      for (int mt = 0; mt < 4; ++mt) {
        const u32x4* p = (const u32x4*)(ap + (size_t)mt * 16 * I_DIM);
        a[mt].u[0] = p[0];        // K = kb + koff .. +7
        a[mt].u[1] = p[2];        // K = kb + 16 + koff .. +23
      }
#pragma unroll
      for (int nt = 0; nt < 2; ++nt) {
        const u32x4* q = (const u32x4*)(bp + (size_t)nt * 16 * I_DIM);
        bf[nt].u[0] = q[0];
        bf[nt].u[1] = q[2];
      }
      ap += 32;                   // next K chunk
      bp += 32;
#pragma unroll
      for (int mt = 0; mt < 4; ++mt)
#pragma unroll
        for (int nt = 0; nt < 2; ++nt)
          acc[mt][nt] = __builtin_amdgcn_wmma_f32_16x16x32_bf16(
              false, a[mt].v, false, bf[nt].v, (short)0, acc[mt][nt],
              false, false);
    }

    // Gate-weighted combine: tot += g[row, e] * acc.
    // C/D layout: element (vgpr v, lane) -> row = v + 8*half, col = l16.
#pragma unroll
    for (int mt = 0; mt < 4; ++mt) {
      const int rowh = rbase + mt * 16 + 8 * half;
#pragma unroll
      for (int v = 0; v < 8; ++v) {
        const float gv = gating[(size_t)(rowh + v) * E_DIM + e];
        tot[mt][0][v] = __fmaf_rn(gv, acc[mt][0][v], tot[mt][0][v]);
        tot[mt][1][v] = __fmaf_rn(gv, acc[mt][1][v], tot[mt][1][v]);
      }
    }
  }

  // Store, adding the gating @ expert_biases term in fp32.
#pragma unroll
  for (int mt = 0; mt < 4; ++mt) {
    const int rowh = rbase + mt * 16 + 8 * half;
#pragma unroll
    for (int nt = 0; nt < 2; ++nt) {
      const int col = cbase + nt * 16 + l16;
#pragma unroll
      for (int v = 0; v < 8; ++v) {
        const int row = rowh + v;
        float val = tot[mt][nt][v];
#pragma unroll
        for (int e = 0; e < E_DIM; ++e)
          val = __fmaf_rn(gating[(size_t)row * E_DIM + e],
                          bias[(size_t)e * O_DIM + col], val);
        out[(size_t)row * O_DIM + col] = val;
      }
    }
  }
}

// ---------------------------------------------------------------------------
extern "C" void kernel_launch(void* const* d_in, const int* in_sizes, int n_in,
                              void* d_out, int out_size, void* d_ws, size_t ws_size,
                              hipStream_t stream) {
  const float* x    = (const float*)d_in[0];   // [B, I]
  const float* cond = (const float*)d_in[1];   // [B, C]
  const float* w    = (const float*)d_in[2];   // [E, I, O]
  const float* bias = (const float*)d_in[3];   // [E, O]
  const float* g_w1 = (const float*)d_in[4];   // [C, 4E]
  const float* g_b1 = (const float*)d_in[5];   // [4E]
  const float* g_w2 = (const float*)d_in[6];   // [4E, E]
  const float* g_b2 = (const float*)d_in[7];   // [E]
  float* out = (float*)d_out;

  // Workspace layout (16B aligned):
  //   gating fp32 [B, E]        : 4096*8*4      =   131072 B
  //   xb  bf16 [B, I]           : 4096*1024*2   =  8388608 B
  //   wt  bf16 [E, O, I]        : 8*1024*1024*2 = 16777216 B
  char* ws = (char*)d_ws;
  float*          gating = (float*)ws;
  unsigned short* xb     = (unsigned short*)(ws + 131072);
  unsigned short* wt     = (unsigned short*)(ws + 131072 + 8388608);

  k_gating<<<B_DIM / 256, 256, 0, stream>>>(cond, g_w1, g_b1, g_w2, g_b2, gating);
  k_cvt_x<<<(B_DIM * I_DIM / 4) / 256, 256, 0, stream>>>(x, xb);
  k_transpose_w<<<dim3(O_DIM / 32, I_DIM / 32, E_DIM), dim3(32, 8), 0, stream>>>(w, wt);
  k_moe_gemm<<<dim3(O_DIM / 128, B_DIM / 128), 256, 0, stream>>>(xb, wt, gating, bias, out);
}
